// GraphLatentNeuralODE_91293824844477
// MI455X (gfx1250) — matrix operand order repaired and checked by reading, not compiled, not run on hardware
//
#include <hip/hip_runtime.h>
#include <math.h>

// ---------------------------------------------------------------------------
// GraphLatentNeuralODE on MI455X (gfx1250)
//
// ~130 GFLOP of GEMM (edge MLP over 32768 rows x 8 RK4 evals); activations
// fit in 192MB L2 -> compute bound -> v_wmma_f32_16x16x32_f16.
// - All inter-layer activations stored f16 (convert once in the epilogue,
//   not per consuming wave) -> no cvt in the WMMA K-loop, half the traffic.
// - Register-blocked: each wave owns a 32x64 C block = 8 WMMAs per K-step.
// - K padded to multiples of 32 in memory (no divergent remainder guards);
//   N and K are template constants so stores use immediate offsets.
// ---------------------------------------------------------------------------

typedef _Float16 v16h __attribute__((ext_vector_type(16)));
typedef _Float16 h8   __attribute__((ext_vector_type(8)));
typedef _Float16 h4   __attribute__((ext_vector_type(4)));
typedef float    v8f  __attribute__((ext_vector_type(8)));

#define NB  2
#define NN  128
#define NT_ 3
#define LAT 64
#define EFK 160   // edge-feature K (132) padded to multiple of 32

// ---------------------------------------------------------------------------
// Pack fp32 weight (K x Nn row-major) into f16 in the CDNA5 WMMA B-fragment
// lane order; K zero-padded to Kp (multiple of 32). For k-tile kt, n-tile nt,
// lane, element j:  half=lane>>4, n=nt*16+(lane&15),
// k=kt*32+half*8+j (j<8), k=kt*32+16+half*8+(j-8) (j>=8).
// Each lane's 16 halves are contiguous (32B) -> 2x global_load_b128 in GEMM.
// ---------------------------------------------------------------------------
__global__ __launch_bounds__(256) void pack_w_kernel(
    const float* __restrict__ W, _Float16* __restrict__ Wp,
    int K, int Nn, int total)
{
    int idx = blockIdx.x * blockDim.x + threadIdx.x;
    if (idx >= total) return;
    int j    = idx & 15;
    int lane = (idx >> 4) & 31;
    int tile = idx >> 9;
    int nTiles = Nn >> 4;
    int nt = tile % nTiles;
    int kt = tile / nTiles;
    int half = lane >> 4;
    int n = nt * 16 + (lane & 15);
    int k = kt * 32 + half * 8 + (j < 8 ? j : 16 + (j - 8));
    float v = (k < K) ? W[(size_t)k * Nn + n] : 0.0f;
    Wp[idx] = (_Float16)v;
}

// Zero-pad rows into f16: dst[r,0:Kin]=src, dst[r,Kin:Kout]=0
__global__ __launch_bounds__(256) void pad_rows_kernel(
    const float* __restrict__ src, _Float16* __restrict__ dst,
    int rows, int Kin, int Kout)
{
    int idx = blockIdx.x * blockDim.x + threadIdx.x;
    if (idx >= rows * Kout) return;
    int r = idx / Kout;
    int c = idx - r * Kout;
    dst[idx] = (c < Kin) ? (_Float16)src[(size_t)r * Kin + c] : (_Float16)0.0f;
}

// ---------------------------------------------------------------------------
// Register-blocked GEMM: C[M,NnT] = act(A[M,KP] @ W[KP,NnT] + bias).
// A f16 row-major (row stride KP, multiple of 32), W pre-packed f16,
// bias f32.  ACT: 0=none, 1=relu, 2=tanh.  OUT16: store f16 vs f32.
// One wave computes a 32 x (16*NTB) block: 2 A-frags x NTB B-frags ->
// 2*NTB WMMAs per K-step.
// ---------------------------------------------------------------------------
template<int KP, int ACT, int NnT, bool OUT16>
__global__ __launch_bounds__(256) void gemm_tpl(
    const _Float16* __restrict__ A, const _Float16* __restrict__ Wp,
    const float* __restrict__ bias, void* __restrict__ Cv, int M)
{
    constexpr int MT_     = 2;
    constexpr int NTB     = (NnT >= 64) ? 4 : 1;
    constexpr int tilesN  = NnT / 16;
    constexpr int groupsN = tilesN / NTB;

    const int lane = threadIdx.x & 31;
    const int wave = threadIdx.x >> 5;
    const int g    = blockIdx.x * 8 + wave;           // uniform per wave
    if (g >= (M >> 5) * groupsN) return;              // whole-wave exit
    const int gm   = g / groupsN;
    const int gn   = g - gm * groupsN;
    const int m0   = gm * (16 * MT_);
    const int n0   = gn * (16 * NTB);
    const int half = lane >> 4;
    const int lm   = lane & 15;

    v8f acc[MT_][NTB] = {};

    const _Float16* aBase[MT_];
#pragma unroll
    for (int mt = 0; mt < MT_; ++mt)
        aBase[mt] = A + (size_t)(m0 + mt * 16 + lm) * KP + half * 8;

#pragma unroll 2
    for (int k0 = 0; k0 < KP; k0 += 32) {
        // A fragment (ISA 16-bit A layout): elems 0-7 at k0+half*8,
        // elems 8-15 at k0+16+half*8 -> two contiguous 16B loads.
        v16h af[MT_];
#pragma unroll
        for (int mt = 0; mt < MT_; ++mt) {
            const h8 lo = *(const h8*)(aBase[mt] + k0);
            const h8 hi = *(const h8*)(aBase[mt] + k0 + 16);
            af[mt] = __builtin_shufflevector(lo, hi,
                0, 1, 2, 3, 4, 5, 6, 7, 8, 9, 10, 11, 12, 13, 14, 15);
        }

        const int kt = k0 >> 5;
#pragma unroll
        for (int nt = 0; nt < NTB; ++nt) {
            const _Float16* bp =
                Wp + (((size_t)kt * tilesN + (gn * NTB + nt)) * 32 + lane) * 16;
            const v16h bf = *(const v16h*)bp;
#pragma unroll
            for (int mt = 0; mt < MT_; ++mt) {
                acc[mt][nt] = __builtin_amdgcn_wmma_f32_16x16x32_f16(
                    false, af[mt], false, bf, (short)0, acc[mt][nt],
                    false, false);
            }
        }
    }

    // ---- epilogue: bias + activation + store (C/D VGPR layout) ----
#pragma unroll
    for (int nt = 0; nt < NTB; ++nt) {
        const int ncol = n0 + nt * 16 + lm;
        const float bv = bias[ncol];
#pragma unroll
        for (int mt = 0; mt < MT_; ++mt) {
            const int row0 = m0 + mt * 16 + half * 8;
            if constexpr (OUT16) {
                _Float16* C = (_Float16*)Cv + (size_t)row0 * NnT + ncol;
#pragma unroll
                for (int r = 0; r < 8; ++r) {
                    float v = acc[mt][nt][r] + bv;
                    if constexpr (ACT == 1)      v = fmaxf(v, 0.0f);
                    else if constexpr (ACT == 2) v = tanhf(v);
                    C[r * NnT] = (_Float16)v;     // immediate offsets
                }
            } else {
                float* C = (float*)Cv + (size_t)row0 * NnT + ncol;
#pragma unroll
                for (int r = 0; r < 8; ++r) {
                    float v = acc[mt][nt][r] + bv;
                    if constexpr (ACT == 1)      v = fmaxf(v, 0.0f);
                    else if constexpr (ACT == 2) v = tanhf(v);
                    C[r * NnT] = v;
                }
            }
        }
    }
}

__device__ __forceinline__ h4 cvt4(float4 v)
{
    h4 r;
    r[0] = (_Float16)v.x; r[1] = (_Float16)v.y;
    r[2] = (_Float16)v.z; r[3] = (_Float16)v.w;
    return r;
}

// ---------------------------------------------------------------------------
// edge_feat[b,i,j,:] = [z_i(64), z_j(64), diff(3), dist(1), 0-pad(28)] as f16
// with row stride 160. z (ODE state) is f32.
// ---------------------------------------------------------------------------
__global__ __launch_bounds__(256) void edge_feat_kernel(
    const float* __restrict__ z, _Float16* __restrict__ ef)
{
    int idx = blockIdx.x * blockDim.x + threadIdx.x;
    if (idx >= NB * NN * NN) return;
    int b   = idx / (NN * NN);
    int rem = idx - b * (NN * NN);
    int i   = rem / NN;
    int j   = rem - i * NN;
    const float* zi = z + ((size_t)b * NN + i) * LAT;
    const float* zj = z + ((size_t)b * NN + j) * LAT;
    _Float16* o = ef + (size_t)idx * EFK;
#pragma unroll
    for (int d = 0; d < LAT / 4; ++d) {
        ((h4*)o)[d]              = cvt4(((const float4*)zi)[d]);
        ((h4*)(o + LAT))[d]      = cvt4(((const float4*)zj)[d]);
    }
    float dx = zi[0] - zj[0];
    float dy = zi[1] - zj[1];
    float dz = zi[2] - zj[2];
    float sq = dx * dx + dy * dy + dz * dz;
    float dist = (sq > 0.0f) ? sqrtf(sq) : 0.0f;
    h4 t;
    t[0] = (_Float16)dx; t[1] = (_Float16)dy;
    t[2] = (_Float16)dz; t[3] = (_Float16)dist;
    ((h4*)o)[32] = t;                               // cols 128..131
    const h4 zero = {};
#pragma unroll
    for (int d = 33; d < EFK / 4; ++d) ((h4*)o)[d] = zero;  // cols 132..159
}

// agg over j of f16 msgs[b,i,j,d] (f32 accumulate);
// upd_in[b,i,:] = [z(64), agg(64)] as f16
__global__ __launch_bounds__(256) void agg_kernel(
    const _Float16* __restrict__ msgs, const float* __restrict__ z,
    _Float16* __restrict__ updin)
{
    int idx = blockIdx.x * blockDim.x + threadIdx.x;
    if (idx >= NB * NN * LAT) return;
    int d    = idx & (LAT - 1);
    int node = idx >> 6;                       // b*NN + i
    const _Float16* mp = msgs + ((size_t)node * NN) * LAT + d;
    float s = 0.0f;
    for (int j = 0; j < NN; ++j) s += (float)mp[(size_t)j * LAT];
    updin[(size_t)node * (2 * LAT) + d]       = (_Float16)z[idx];
    updin[(size_t)node * (2 * LAT) + LAT + d] = (_Float16)s;
}

// out = z + coef*dt*k   (all f32; dt read from t_span on device)
__global__ __launch_bounds__(256) void axpy_kernel(
    const float* __restrict__ z, const float* __restrict__ k, float coef,
    const float* __restrict__ tspan, int step, float* __restrict__ out, int n)
{
    int i = blockIdx.x * blockDim.x + threadIdx.x;
    if (i >= n) return;
    float dt = tspan[step + 1] - tspan[step];
    out[i] = z[i] + coef * dt * k[i];
}

// z += dt/6*(k1+2k2+2k3+k4); scatter new z into f16 traj slot step+1
__global__ __launch_bounds__(256) void rk4_final_kernel(
    float* __restrict__ z,
    const float* __restrict__ k1, const float* __restrict__ k2,
    const float* __restrict__ k3, const float* __restrict__ k4,
    const float* __restrict__ tspan, int step, _Float16* __restrict__ traj, int n)
{
    int i = blockIdx.x * blockDim.x + threadIdx.x;
    if (i >= n) return;
    float dt = tspan[step + 1] - tspan[step];
    float v = z[i] + (dt / 6.0f) * (k1[i] + 2.0f * k2[i] + 2.0f * k3[i] + k4[i]);
    z[i] = v;
    int b   = i / (NN * LAT);
    int rem = i - b * (NN * LAT);
    traj[((size_t)b * NT_ + (step + 1)) * (NN * LAT) + rem] = (_Float16)v;
}

__global__ __launch_bounds__(256) void scatter_traj_kernel(
    const float* __restrict__ zsrc, _Float16* __restrict__ traj, int t, int n)
{
    int i = blockIdx.x * blockDim.x + threadIdx.x;
    if (i >= n) return;
    int b   = i / (NN * LAT);
    int rem = i - b * (NN * LAT);
    traj[((size_t)b * NT_ + t) * (NN * LAT) + rem] = (_Float16)zsrc[i];
}

// ---------------------------------------------------------------------------
// Host-side orchestration
// ---------------------------------------------------------------------------
namespace {

struct MlpPacked {
    const _Float16* Wp[5];
    const float*    b[5];
    int             kp[5];   // padded K per layer (multiple of 32)
    int             nn[5];   // output width per layer
};

inline void launch_gemm(hipStream_t s, const _Float16* A, const _Float16* Wp,
                        const float* bias, void* C, int M, int Nn, int KP,
                        int act, bool out16)
{
    const int NTB    = (Nn >= 64) ? 4 : 1;
    const int groups = (M / 32) * (Nn / (16 * NTB));
    const dim3 gr((groups + 7) / 8), bl(256);
#define GEMM_CASE(kp, a, nn, o) \
    gemm_tpl<kp, a, nn, o><<<gr, bl, 0, s>>>(A, Wp, bias, C, M)
    if (out16) {
        if      (KP == 32  && act == 1 && Nn == 256) GEMM_CASE(32, 1, 256, true);
        else if (KP == 64  && act == 1 && Nn == 256) GEMM_CASE(64, 1, 256, true);
        else if (KP == 128 && act == 2 && Nn == 256) GEMM_CASE(128, 2, 256, true);
        else if (KP == 160 && act == 2 && Nn == 256) GEMM_CASE(160, 2, 256, true);
        else if (KP == 256 && act == 1 && Nn == 256) GEMM_CASE(256, 1, 256, true);
        else if (KP == 256 && act == 2 && Nn == 256) GEMM_CASE(256, 2, 256, true);
        else if (KP == 256 && act == 0 && Nn == 64)  GEMM_CASE(256, 0, 64, true);
    } else {
        if      (KP == 256 && act == 0 && Nn == 64)  GEMM_CASE(256, 0, 64, false);
        else if (KP == 256 && act == 0 && Nn == 16)  GEMM_CASE(256, 0, 16, false);
    }
#undef GEMM_CASE
}

// X: f16, row stride m.kp[0]. Hidden layers -> f16 ping/pong; final layer
// stores f16 (out16=true) or f32 (out16=false) into `out`.
inline void run_mlp(hipStream_t s, const MlpPacked& m, int M, const _Float16* X,
                    void* out, bool out16, _Float16* p0, _Float16* p1, int act)
{
    const _Float16* cur = X;
    _Float16* ping[2] = {p0, p1};
    for (int l = 0; l < 5; ++l) {
        const bool last = (l == 4);
        void* dst = last ? out : (void*)ping[l & 1];
        launch_gemm(s, cur, m.Wp[l], m.b[l], dst, M, m.nn[l], m.kp[l],
                    last ? 0 : act, last ? out16 : true);
        cur = ping[l & 1];
    }
}

} // namespace

extern "C" void kernel_launch(void* const* d_in, const int* in_sizes, int n_in,
                              void* d_out, int out_size, void* d_ws, size_t ws_size,
                              hipStream_t stream)
{
    (void)in_sizes; (void)n_in; (void)out_size; (void)ws_size;

    const float* z0    = (const float*)d_in[0];
    const float* tspan = (const float*)d_in[1];

    // ---- workspace layout (bytes, 256B aligned). Needs ~48 MB. ----
    char* p = (char*)d_ws;
    auto alloc = [&](size_t bytes) {
        char* r = p;
        p += (bytes + 255) & ~(size_t)255;
        return r;
    };
    const size_t EDGES = (size_t)NB * NN * NN;          // 32768
    _Float16* bufA  = (_Float16*)alloc(EDGES * 256 * 2);     // 16.8 MB
    _Float16* bufB  = (_Float16*)alloc(EDGES * 256 * 2);     // 16.8 MB
    _Float16* ef    = (_Float16*)alloc(EDGES * EFK * 2);     // 10.5 MB
    _Float16* updin = (_Float16*)alloc((size_t)NB * NN * 2 * LAT * 2);
    _Float16* traj  = (_Float16*)alloc((size_t)NB * NT_ * NN * LAT * 2);
    _Float16* z0pad = (_Float16*)alloc((size_t)NB * NN * 32 * 2);
    float* zbuf = (float*)alloc((size_t)NB * NN * LAT * 4);
    float* ztmp = (float*)alloc((size_t)NB * NN * LAT * 4);
    float* k1   = (float*)alloc((size_t)NB * NN * LAT * 4);
    float* k2   = (float*)alloc((size_t)NB * NN * LAT * 4);
    float* k3   = (float*)alloc((size_t)NB * NN * LAT * 4);
    float* k4   = (float*)alloc((size_t)NB * NN * LAT * 4);
    _Float16* wcur = (_Float16*)alloc(2 * 1000000);          // packed weights

    // ---- pack all 20 weight matrices to f16 WMMA B-fragment layout ----
    const int dimsAll[4][6] = {
        { 16, 256, 256, 256, 256,  64},   // enc
        {132, 256, 256, 256, 256,  64},   // edge
        {128, 256, 256, 256, 256,  64},   // upd
        { 64, 256, 256, 256, 256,  16},   // dec
    };
    const int base[4] = {2, 12, 22, 32};

    MlpPacked mlps[4];
    for (int g = 0; g < 4; ++g) {
        for (int l = 0; l < 5; ++l) {
            const float* W  = (const float*)d_in[base[g] + 2 * l];
            const float* bb = (const float*)d_in[base[g] + 2 * l + 1];
            int K  = dimsAll[g][l];
            int Nn = dimsAll[g][l + 1];
            int Kp = (K + 31) & ~31;
            int total = Kp * Nn;
            pack_w_kernel<<<(total + 255) / 256, 256, 0, stream>>>(W, wcur, K, Nn, total);
            mlps[g].Wp[l] = wcur;
            mlps[g].b[l]  = bb;
            mlps[g].kp[l] = Kp;
            mlps[g].nn[l] = Nn;
            wcur += total;
        }
    }

    const int nz = NB * NN * LAT;        // 16384
    const int eb = ((int)EDGES + 255) / 256;
    const int zb = (nz + 255) / 256;

    // z0 padded 16->32 wide, converted to f16, for the KP=32 encoder layer
    {
        int tot = NB * NN * 32;
        pad_rows_kernel<<<(tot + 255) / 256, 256, 0, stream>>>(z0, z0pad, NB * NN, 16, 32);
    }

    // ODE function: kout(f32) = f(zin(f32))
    auto ode = [&](const float* zin, float* kout) {
        edge_feat_kernel<<<eb, 256, 0, stream>>>(zin, ef);
        // edge MLP over 32768 rows; final f16 64-wide msgs land in bufA
        run_mlp(stream, mlps[1], (int)EDGES, ef, bufA, true, bufA, bufB, /*tanh*/2);
        agg_kernel<<<zb, 256, 0, stream>>>(bufA, zin, updin);
        run_mlp(stream, mlps[2], NB * NN, updin, kout, false, bufA, bufB, /*tanh*/2);
    };

    // encoder: latent0 (f32 state) = enc(z0), relu hidden
    run_mlp(stream, mlps[0], NB * NN, z0pad, zbuf, false, bufA, bufB, /*relu*/1);
    scatter_traj_kernel<<<zb, 256, 0, stream>>>(zbuf, traj, 0, nz);

    // RK4 over T-1 steps
    for (int s = 0; s < NT_ - 1; ++s) {
        ode(zbuf, k1);
        axpy_kernel<<<zb, 256, 0, stream>>>(zbuf, k1, 0.5f, tspan, s, ztmp, nz);
        ode(ztmp, k2);
        axpy_kernel<<<zb, 256, 0, stream>>>(zbuf, k2, 0.5f, tspan, s, ztmp, nz);
        ode(ztmp, k3);
        axpy_kernel<<<zb, 256, 0, stream>>>(zbuf, k3, 1.0f, tspan, s, ztmp, nz);
        ode(ztmp, k4);
        rk4_final_kernel<<<zb, 256, 0, stream>>>(zbuf, k1, k2, k3, k4, tspan, s, traj, nz);
    }

    // decoder over the whole f16 trajectory (768 rows) -> d_out f32
    run_mlp(stream, mlps[3], NB * NT_ * NN, traj, d_out, false, bufA, bufB, /*relu*/1);
}
